// MultiHeadSelfAttention_72499047957110
// MI455X (gfx1250) — matrix operand order repaired and checked by reading, not compiled, not run on hardware
//
#include <hip/hip_runtime.h>
#include <hip/hip_bf16.h>

typedef __attribute__((ext_vector_type(16))) __bf16 v16bf;
typedef __attribute__((ext_vector_type(8)))  float  v8f;

constexpr int Bc = 4, Sc = 2048, Dc = 1024, Hc = 16, HDc = 64;

// ---------------------------------------------------------------------------
// Tensor Data Mover: 2D tile global->LDS with row padding (D# per ISA 8.3/8.4)
// ---------------------------------------------------------------------------
#if __has_builtin(__builtin_amdgcn_tensor_load_to_lds)
#define HAS_TDM 1
#else
#define HAS_TDM 0
#endif

#if HAS_TDM
typedef __attribute__((ext_vector_type(4))) unsigned int v4u;
typedef __attribute__((ext_vector_type(8))) int v8i;
typedef __attribute__((ext_vector_type(4))) int v4i;

// Copies tileH rows x tileW bf16 elements from global (row stride
// rowStrideElems) into LDS at ldst, inserting pad after each row:
//   padIntervalCode: (1<<code) 8-byte units per row payload
//   padAmountCode:   (code+1)-ish dwords of LDS padding per row
__device__ __forceinline__ void tdm_load_2d_to_lds(
    const void* gsrc, void* ldst,
    unsigned tileW, unsigned tileH, unsigned rowStrideElems,
    unsigned padIntervalCode, unsigned padAmountCode)
{
    unsigned long long ga = (unsigned long long)gsrc;
    v4u g0 = { 1u,                                          // count=1 (valid D#)
               (unsigned)(unsigned long long)ldst,          // lds byte address
               (unsigned)ga,                                // global addr [31:0]
               (unsigned)((ga >> 32) & 0x01ffffffu) | (2u << 30) }; // [56:32]|type=2
    unsigned dw0 = (1u << 16)                               // data_size = 2 bytes
                 | (1u << 20)                               // pad_enable
                 | (padIntervalCode << 22)
                 | (padAmountCode << 25);
    unsigned td0 = rowStrideElems;                          // tensor_dim0
    unsigned td1 = tileH;                                   // tensor_dim1
    v8i g1 = {
        (int)dw0,
        (int)((td0 & 0xffffu) << 16),                       // dim0 lo16 (atomic addr=0)
        (int)((td0 >> 16) | ((td1 & 0xffffu) << 16)),       // dim0 hi | dim1 lo
        (int)((td1 >> 16) | (tileW << 16)),                 // dim1 hi | tile_dim0
        (int)(tileH & 0xffffu),                             // tile_dim1 | tile_dim2=0
        (int)rowStrideElems,                                // tensor_dim0_stride lo32
        0, 0 };
    v4i z4 = {0, 0, 0, 0};
#if defined(__clang_major__) && __clang_major__ >= 23
    v8i z8 = {0, 0, 0, 0, 0, 0, 0, 0};
    __builtin_amdgcn_tensor_load_to_lds(g0, g1, z4, z4, z8, 0);
#else
    __builtin_amdgcn_tensor_load_to_lds(g0, g1, z4, z4, 0);
#endif
}
#endif

// ---------------------------------------------------------------------------
// f32 -> bf16 conversion (grid-stride)
// ---------------------------------------------------------------------------
__global__ void cvt_bf16_kernel(const float* __restrict__ in,
                                __bf16* __restrict__ out, int n) {
    int i = blockIdx.x * blockDim.x + threadIdx.x;
    int stride = gridDim.x * blockDim.x;
    for (; i < n; i += stride) out[i] = (__bf16)in[i];
}

// ---------------------------------------------------------------------------
// Tiled bf16 GEMM, f32 accumulate:  C = (A @ B + bias) * scale
//   A: [M,K] bf16 row-major, B: [K,N] bf16 row-major, bias non-null.
//   OUTF32: epilogue writes f32, else bf16.
// Block: 256 threads (8 waves), 128x128 tile; wave tile 64x32; K-step 32.
// ---------------------------------------------------------------------------
template <bool OUTF32>
__global__ __launch_bounds__(256) void gemm_bf16_kernel(
    const __bf16* __restrict__ A, const __bf16* __restrict__ Bw,
    const float* __restrict__ bias, float scale,
    void* __restrict__ outPtr, int M, int N, int K)
{
    __shared__ __align__(16) __bf16 As[128][40];   // 128x32 + 8 pad
    __shared__ __align__(16) __bf16 BsT[128][40];  // B transposed [n][k] + pad

    const unsigned tid  = threadIdx.x;
    const unsigned wave = tid >> 5, lane = tid & 31;
    const unsigned half = lane >> 4, l15 = lane & 15;
    const unsigned waveM = wave >> 2, waveN = wave & 3;   // 2x4 wave grid
    const unsigned blockM = blockIdx.y * 128;
    const unsigned blockN = blockIdx.x * 128;
    const unsigned uN = (unsigned)N, uK = (unsigned)K;

    v8f acc[4][2] = {};

    for (unsigned k0 = 0; k0 < uK; k0 += 32) {
        // --- A tile: 128x32 ---
#if HAS_TDM
        if (wave == 0) {
            tdm_load_2d_to_lds(A + blockM * uK + k0, &As[0][0],
                               /*tileW=*/32, /*tileH=*/128, /*stride=*/uK,
                               /*padIntervalCode=*/3 /*64B rows*/,
                               /*padAmountCode=*/3 /*16B pad*/);
            __builtin_amdgcn_s_wait_tensorcnt(0);
        }
#else
        {
            unsigned r = tid >> 1;
            unsigned c = (tid & 1) * 16;
            const uint4* src = (const uint4*)(A + (blockM + r) * uK + k0 + c);
            *(uint4*)(&As[r][c])     = src[0];
            *(uint4*)(&As[r][c + 8]) = src[1];
        }
#endif
        // --- B tile: 32x128, store transposed ---
        {
            unsigned kr   = tid >> 3;
            unsigned nOff = (tid & 7) * 16;
            const __bf16* src = Bw + (k0 + kr) * uN + blockN + nOff;
            __bf16 tmp[16];
            *(uint4*)&tmp[0] = *(const uint4*)(src);
            *(uint4*)&tmp[8] = *(const uint4*)(src + 8);
#pragma unroll
            for (int e = 0; e < 16; ++e) BsT[nOff + e][kr] = tmp[e];
        }
        // prefetch next K-step tiles into L2 while this step computes
        if (k0 + 32 < uK) {
            __builtin_prefetch(A + (blockM + (tid >> 1)) * uK + k0 + 32, 0, 0);
            __builtin_prefetch(Bw + (k0 + 32 + (tid >> 3)) * uN + blockN + (tid & 7) * 16, 0, 0);
        }
        __syncthreads();

        v16bf aF[4], bF[2];
        const unsigned kb = half * 8;
#pragma unroll
        for (int mi = 0; mi < 4; ++mi) {
            unsigned rowA = waveM * 64 + mi * 16 + l15;
            ((uint4*)&aF[mi])[0] = *(const uint4*)(&As[rowA][kb]);
            ((uint4*)&aF[mi])[1] = *(const uint4*)(&As[rowA][kb + 16]);
        }
        const unsigned kbB = half * 8;
#pragma unroll
        for (int ni = 0; ni < 2; ++ni) {
            unsigned colB = waveN * 32 + ni * 16 + l15;
            ((uint4*)&bF[ni])[0] = *(const uint4*)(&BsT[colB][kbB]);
            ((uint4*)&bF[ni])[1] = *(const uint4*)(&BsT[colB][kbB + 16]);
        }
#pragma unroll
        for (int mi = 0; mi < 4; ++mi)
#pragma unroll
            for (int ni = 0; ni < 2; ++ni)
                acc[mi][ni] = __builtin_amdgcn_wmma_f32_16x16x32_bf16(
                    false, aF[mi], false, bF[ni], (short)0, acc[mi][ni], false, false);
        __syncthreads();
    }

    // --- epilogue (branch-free, 32-bit addressing) ---
    float* outF  = (float*)outPtr;
    __bf16* outB = (__bf16*)outPtr;
#pragma unroll
    for (int ni = 0; ni < 2; ++ni) {
        unsigned col = blockN + waveN * 32 + ni * 16 + l15;
        float bv = bias[col];
#pragma unroll
        for (int mi = 0; mi < 4; ++mi) {
            unsigned rowBase = blockM + waveM * 64 + mi * 16 + half * 8;
#pragma unroll
            for (int i = 0; i < 8; ++i) {
                unsigned idx = (rowBase + i) * uN + col;
                float v = (acc[mi][ni][i] + bv) * scale;
                if constexpr (OUTF32) outF[idx] = v;
                else                  outB[idx] = (__bf16)v;
            }
        }
    }
}

// ---------------------------------------------------------------------------
// Flash attention: one block per (b, h, 64-query tile); 4 waves x 16 rows.
// Q/K/V: bf16 [B*S, D] with head h at columns h*64..h*64+63. Q pre-scaled.
// Ctx out: bf16 [B*S, D] (merged heads).
// ---------------------------------------------------------------------------
__global__ __launch_bounds__(128) void flash_attn_kernel(
    const __bf16* __restrict__ Q, const __bf16* __restrict__ Kg,
    const __bf16* __restrict__ Vg, __bf16* __restrict__ Ctx)
{
    __shared__ __align__(16) __bf16 Kt[64][72];      // [key][d] (+8 pad)
    __shared__ __align__(16) __bf16 Vt[64][72];      // [d][key] (transposed)
    __shared__ __align__(16) __bf16 Pb[4][16][72];   // per-wave P re-layout

    const unsigned tid  = threadIdx.x;
    const unsigned wave = tid >> 5, lane = tid & 31;
    const unsigned half = lane >> 4, l15 = lane & 15;

    const unsigned qtPerSeq = Sc / 64;     // 32
    const unsigned bid = blockIdx.x;
    const unsigned qt = bid % qtPerSeq;
    const unsigned h  = (bid / qtPerSeq) % Hc;
    const unsigned b  = bid / (qtPerSeq * Hc);

    const unsigned qRowBase = qt * 64 + wave * 16;
    const unsigned seqBase = b * Sc;

    // Q fragments (held in registers for the whole KV loop)
    v16bf Qa[2];
    {
        unsigned qRow = qRowBase + l15;
        const __bf16* qp = Q + (seqBase + qRow) * Dc + h * HDc;
        unsigned kb = half * 8;
#pragma unroll
        for (int s = 0; s < 2; ++s) {
            ((uint4*)&Qa[s])[0] = *(const uint4*)(qp + s * 32 + kb);
            ((uint4*)&Qa[s])[1] = *(const uint4*)(qp + s * 32 + kb + 16);
        }
    }

    v8f Oacc[4] = {};
    float rmax[8], rsum[8];
#pragma unroll
    for (int i = 0; i < 8; ++i) { rmax[i] = -3.0e38f; rsum[i] = 0.f; }

    for (unsigned ktile = 0; ktile < Sc / 64; ++ktile) {
        const unsigned keyRow0 = seqBase + ktile * 64;
        // --- K tile (row-major): TDM DMA if available ---
#if HAS_TDM
        if (wave == 0) {
            tdm_load_2d_to_lds(Kg + keyRow0 * Dc + h * HDc, &Kt[0][0],
                               /*tileW=*/64, /*tileH=*/64, /*stride=*/Dc,
                               /*padIntervalCode=*/4 /*128B rows*/,
                               /*padAmountCode=*/3 /*16B pad*/);
            __builtin_amdgcn_s_wait_tensorcnt(0);
        }
#else
        {
            unsigned r = tid >> 1;
            unsigned c = (tid & 1) * 32;
            const __bf16* kp = Kg + (keyRow0 + r) * Dc + h * HDc + c;
#pragma unroll
            for (int e = 0; e < 4; ++e)
                *(uint4*)(&Kt[r][c + e * 8]) = *(const uint4*)(kp + e * 8);
        }
#endif
        // --- V tile, transposed (all threads) ---
        {
            unsigned r = tid >> 1;
            unsigned c = (tid & 1) * 32;
            const __bf16* vp = Vg + (keyRow0 + r) * Dc + h * HDc + c;
            __bf16 tv[32];
#pragma unroll
            for (int e = 0; e < 4; ++e)
                *(uint4*)(&tv[e * 8]) = *(const uint4*)(vp + e * 8);
#pragma unroll
            for (int e = 0; e < 32; ++e) Vt[c + e][r] = tv[e];
        }
        if (ktile + 1 < Sc / 64) {
            unsigned r = tid >> 1;
            unsigned c = (tid & 1) * 32;
            __builtin_prefetch(Kg + (keyRow0 + 64 + r) * Dc + h * HDc + c, 0, 0);
            __builtin_prefetch(Vg + (keyRow0 + 64 + r) * Dc + h * HDc + c, 0, 0);
        }
        __syncthreads();

        // S = Q @ K^T  (already scaled via Q)
        v8f Sacc[4] = {};
#pragma unroll
        for (int j = 0; j < 4; ++j) {
#pragma unroll
            for (int s = 0; s < 2; ++s) {
                v16bf bfrag;
                unsigned keyN = j * 16 + l15;
                unsigned d0 = s * 32 + half * 8;
                ((uint4*)&bfrag)[0] = *(const uint4*)(&Kt[keyN][d0]);
                ((uint4*)&bfrag)[1] = *(const uint4*)(&Kt[keyN][d0 + 16]);
                Sacc[j] = __builtin_amdgcn_wmma_f32_16x16x32_bf16(
                    false, Qa[s], false, bfrag, (short)0, Sacc[j], false, false);
            }
        }

        // online softmax (row m = i + 8*half; cols spread over 16-lane half)
#pragma unroll
        for (int i = 0; i < 8; ++i) {
            float mx = fmaxf(fmaxf(Sacc[0][i], Sacc[1][i]),
                             fmaxf(Sacc[2][i], Sacc[3][i]));
#pragma unroll
            for (int msk = 1; msk < 16; msk <<= 1)
                mx = fmaxf(mx, __shfl_xor(mx, msk, 32));
            float nm = fmaxf(rmax[i], mx);
            float corr = __expf(rmax[i] - nm);
            rmax[i] = nm;
            float ps = 0.f;
#pragma unroll
            for (int j = 0; j < 4; ++j) {
                float p = __expf(Sacc[j][i] - nm);
                Sacc[j][i] = p;
                ps += p;
            }
#pragma unroll
            for (int msk = 1; msk < 16; msk <<= 1)
                ps += __shfl_xor(ps, msk, 32);
            rsum[i] = rsum[i] * corr + ps;
#pragma unroll
            for (int j = 0; j < 4; ++j) Oacc[j][i] *= corr;
        }

        // C-layout -> A-layout via per-wave LDS bounce
#pragma unroll
        for (int j = 0; j < 4; ++j)
#pragma unroll
            for (int i = 0; i < 8; ++i)
                Pb[wave][half * 8 + i][j * 16 + l15] = (__bf16)Sacc[j][i];

        asm volatile("s_wait_dscnt 0" ::: "memory");

        v16bf Pa[2];
        {
            const __bf16* pr = &Pb[wave][l15][0];
            unsigned kb = half * 8;
#pragma unroll
            for (int s = 0; s < 2; ++s) {
                ((uint4*)&Pa[s])[0] = *(const uint4*)(pr + s * 32 + kb);
                ((uint4*)&Pa[s])[1] = *(const uint4*)(pr + s * 32 + kb + 16);
            }
        }

        // O += P @ V
#pragma unroll
        for (int j = 0; j < 4; ++j) {
#pragma unroll
            for (int s = 0; s < 2; ++s) {
                v16bf bfrag;
                unsigned drow = j * 16 + l15;
                unsigned key0 = s * 32 + half * 8;
                ((uint4*)&bfrag)[0] = *(const uint4*)(&Vt[drow][key0]);
                ((uint4*)&bfrag)[1] = *(const uint4*)(&Vt[drow][key0 + 16]);
                Oacc[j] = __builtin_amdgcn_wmma_f32_16x16x32_bf16(
                    false, Pa[s], false, bfrag, (short)0, Oacc[j], false, false);
            }
        }
        __syncthreads();
    }

    // normalize + write context (merged-head layout [B*S, D])
    float inv[8];
#pragma unroll
    for (int i = 0; i < 8; ++i) inv[i] = 1.0f / rsum[i];
#pragma unroll
    for (int j = 0; j < 4; ++j) {
        unsigned col = h * HDc + j * 16 + l15;
#pragma unroll
        for (int i = 0; i < 8; ++i) {
            unsigned row = qRowBase + half * 8 + i;
            Ctx[(seqBase + row) * Dc + col] = (__bf16)(Oacc[j][i] * inv[i]);
        }
    }
}

// ---------------------------------------------------------------------------
// host launcher
// ---------------------------------------------------------------------------
extern "C" void kernel_launch(void* const* d_in, const int* in_sizes, int n_in,
                              void* d_out, int out_size, void* d_ws, size_t ws_size,
                              hipStream_t stream) {
    const float* x  = (const float*)d_in[0];
    const float* wq = (const float*)d_in[1];
    const float* bq = (const float*)d_in[2];
    const float* wk = (const float*)d_in[3];
    const float* bk = (const float*)d_in[4];
    const float* wv = (const float*)d_in[5];
    const float* bv = (const float*)d_in[6];
    const float* wo = (const float*)d_in[7];
    const float* bo = (const float*)d_in[8];
    float* out = (float*)d_out;

    char* ws = (char*)d_ws;
    const size_t MB = 1ull << 20;
    __bf16* xb  = (__bf16*)(ws + 0);        // 16 MB, recycled as Ctx later
    __bf16* wqb = (__bf16*)(ws + 16 * MB);  // 2 MB each
    __bf16* wkb = (__bf16*)(ws + 18 * MB);
    __bf16* wvb = (__bf16*)(ws + 20 * MB);
    __bf16* wob = (__bf16*)(ws + 22 * MB);
    __bf16* Qb  = (__bf16*)(ws + 24 * MB);  // 16 MB each
    __bf16* Kb  = (__bf16*)(ws + 40 * MB);
    __bf16* Vb  = (__bf16*)(ws + 56 * MB);
    __bf16* Cb  = xb;                        // context reuses x_bf16 region

    const int M = Bc * Sc;                   // 8192
    const int nx = M * Dc;                   // 8388608
    const int nw = Dc * Dc;                  // 1048576

    cvt_bf16_kernel<<<1024, 256, 0, stream>>>(x,  xb,  nx);
    cvt_bf16_kernel<<<512,  256, 0, stream>>>(wq, wqb, nw);
    cvt_bf16_kernel<<<512,  256, 0, stream>>>(wk, wkb, nw);
    cvt_bf16_kernel<<<512,  256, 0, stream>>>(wv, wvb, nw);
    cvt_bf16_kernel<<<512,  256, 0, stream>>>(wo, wob, nw);

    dim3 gg(Dc / 128, M / 128);              // (8, 64)
    const float qscale = 0.125f;             // 1/sqrt(HD), folded into Q
    gemm_bf16_kernel<false><<<gg, 256, 0, stream>>>(xb, wqb, bq, qscale, (void*)Qb, M, Dc, Dc);
    gemm_bf16_kernel<false><<<gg, 256, 0, stream>>>(xb, wkb, bk, 1.0f,   (void*)Kb, M, Dc, Dc);
    gemm_bf16_kernel<false><<<gg, 256, 0, stream>>>(xb, wvb, bv, 1.0f,   (void*)Vb, M, Dc, Dc);

    flash_attn_kernel<<<Bc * Hc * (Sc / 64), 128, 0, stream>>>(Qb, Kb, Vb, Cb);

    gemm_bf16_kernel<true><<<gg, 256, 0, stream>>>(Cb, wob, bo, 1.0f, (void*)out, M, Dc, Dc);
}
